// MultiHeadedAddAttention_59983513255986
// MI455X (gfx1250) — matrix operand order repaired and checked by reading, not compiled, not run on hardware
//
#include <hip/hip_runtime.h>

typedef __attribute__((ext_vector_type(16))) _Float16 v16h;
typedef __attribute__((ext_vector_type(8)))  float    v8f;

#define D_MODEL 512
#define NHEAD   8
#define DKH     64
#define BATCH   2
#define LQ      256
#define LKV     512
#define KTILES  16   // K = 512 -> 16 steps of 32

// ---------------------------------------------------------------------------
// Pack an MxK f32 row-major matrix into WMMA A-operand layout (f16), tiled
// 16x32. Dest element ((mt*Kt + kt)*32 + lane)*16 + e holds
// A[mt*16 + (lane&15)][kt*32 + kl],
//   kl = 16*(lane>>4) + e
// ---------------------------------------------------------------------------
__global__ void pack_a_kernel(const float* __restrict__ src,
                              _Float16* __restrict__ dst, int M, int K) {
  int tid = blockIdx.x * blockDim.x + threadIdx.x;
  if (tid >= M * K) return;
  int Kt      = K >> 5;
  int e       = tid & 15;
  int lane    = (tid >> 4) & 31;
  int tileIdx = tid >> 9;
  int kt      = tileIdx % Kt;
  int mt      = tileIdx / Kt;
  int half_   = lane >> 4;
  int m       = lane & 15;
  int kl      = 16 * half_ + e;
  dst[tid] = (_Float16)src[(size_t)(mt * 16 + m) * K + kt * 32 + kl];
}

// ---------------------------------------------------------------------------
// Pack a KxN f32 row-major matrix into WMMA B-operand layout (f16), tiled
// 32x16. Dest ((kt*Nt + nt)*32 + lane)*16 + e holds
// B[kt*32 + 16*(lane>>4) + e][nt*16 + (lane&15)]
// ---------------------------------------------------------------------------
__global__ void pack_b_kernel(const float* __restrict__ src,
                              _Float16* __restrict__ dst, int K, int N) {
  int tid = blockIdx.x * blockDim.x + threadIdx.x;
  if (tid >= K * N) return;
  int Nt      = N >> 4;
  int e       = tid & 15;
  int lane    = (tid >> 4) & 31;
  int tileIdx = tid >> 9;
  int nt      = tileIdx % Nt;
  int kt      = tileIdx / Nt;
  int kl      = 16 * (lane >> 4) + e;
  dst[tid] = (_Float16)src[(size_t)(kt * 32 + kl) * N + nt * 16 + (lane & 15)];
}

// ---------------------------------------------------------------------------
// Swizzle row-major f16 V projection (B*Lk x 512) into per-(b,h) B-operand
// layout for the PV matmul:  dst[((((b*H+h)*16 + kt)*4 + nt)*32 + lane)*16+e]
//   = V[b*Lk + kt*32 + 16*(lane>>4) + e][h*64 + nt*16 + (lane&15)]
// ---------------------------------------------------------------------------
__global__ void pack_vb_kernel(const _Float16* __restrict__ src,
                               _Float16* __restrict__ dst) {
  int tid = blockIdx.x * blockDim.x + threadIdx.x;
  if (tid >= BATCH * NHEAD * 16 * 4 * 32 * 16) return;
  int e    = tid & 15;
  int lane = (tid >> 4) & 31;
  int rest = tid >> 9;
  int nt   = rest & 3;
  int kt   = (rest >> 2) & 15;
  int h    = (rest >> 6) & 7;
  int b    = rest >> 9;
  int j    = kt * 32 + 16 * (lane >> 4) + e;
  int col  = h * DKH + nt * 16 + (lane & 15);
  dst[tid] = src[(size_t)(b * LKV + j) * D_MODEL + col];
}

// ---------------------------------------------------------------------------
// C[M][N] f32 = A(f16, packed) x B(f16, packed).
// One wave computes a 16x64 strip (4 adjacent 16x16 tiles): A loaded once per
// K-step, reused by 4 WMMAs. K-loop fully unrolled; per-lane base pointers +
// compile-time byte offsets -> global_load_b128 with immediate offsets.
// ---------------------------------------------------------------------------
__global__ void gemm_wmma_kernel(const _Float16* __restrict__ Ap,
                                 const _Float16* __restrict__ Bp,
                                 int Mt, int Ntg /* N/64 groups */,
                                 float* __restrict__ Cf32,
                                 _Float16* __restrict__ Cf16, int N) {
  const int Nt   = Ntg * 4;                       // 16-col tiles
  const int wave = threadIdx.x >> 5;
  const int lane = threadIdx.x & 31;
  const int strip = blockIdx.x * (blockDim.x >> 5) + wave;
  if (strip >= Mt * Ntg) return;                  // wave-uniform guard
  const int mt  = strip / Ntg;
  const int ntg = strip % Ntg;

  // per-lane base pointers; per-kt strides are compile-time constants
  const _Float16* aB = Ap + ((size_t)mt * KTILES * 32 + lane) * 16;
  const _Float16* b0 = Bp + ((size_t)(ntg * 4 + 0) * 32 + lane) * 16;
  const _Float16* b1 = Bp + ((size_t)(ntg * 4 + 1) * 32 + lane) * 16;
  const _Float16* b2 = Bp + ((size_t)(ntg * 4 + 2) * 32 + lane) * 16;
  const _Float16* b3 = Bp + ((size_t)(ntg * 4 + 3) * 32 + lane) * 16;
  const size_t aStride = 512;                     // halfs per kt (32 lanes*16)
  const size_t bStride = (size_t)Nt * 512;        // halfs per kt

  __builtin_prefetch(aB, 0, 0);
  __builtin_prefetch(b0, 0, 0);

  v8f c0 = {}, c1 = {}, c2 = {}, c3 = {};
#pragma unroll
  for (int kt = 0; kt < KTILES; ++kt) {
    v16h a  = *(const v16h*)(aB + kt * aStride);
    v16h vb0 = *(const v16h*)(b0 + kt * bStride);
    v16h vb1 = *(const v16h*)(b1 + kt * bStride);
    v16h vb2 = *(const v16h*)(b2 + kt * bStride);
    v16h vb3 = *(const v16h*)(b3 + kt * bStride);
    c0 = __builtin_amdgcn_wmma_f32_16x16x32_f16(false, a, false, vb0, (short)0, c0, false, false);
    c1 = __builtin_amdgcn_wmma_f32_16x16x32_f16(false, a, false, vb1, (short)0, c1, false, false);
    c2 = __builtin_amdgcn_wmma_f32_16x16x32_f16(false, a, false, vb2, (short)0, c2, false, false);
    c3 = __builtin_amdgcn_wmma_f32_16x16x32_f16(false, a, false, vb3, (short)0, c3, false, false);
  }

  const int half_ = lane >> 4;
  const int nl    = lane & 15;
  const v8f  cc[4] = {c0, c1, c2, c3};
#pragma unroll
  for (int j = 0; j < 4; ++j) {
    int n = (ntg * 4 + j) * 16 + nl;
#pragma unroll
    for (int r = 0; r < 8; ++r) {                 // VGPR r -> row 8*half + r
      int m = mt * 16 + 8 * half_ + r;
      if (Cf32) Cf32[(size_t)m * N + n] = cc[j][r];
      if (Cf16) Cf16[(size_t)m * N + n] = (_Float16)cc[j][r];
    }
  }
}

// ---------------------------------------------------------------------------
// Additive attention core: one wave per (b, h, 16-query block).
//   scores[i][j] = sum_d tanh(q[i][d] + k[j][d]) * wa[d]  (mask -> -1e9)
//   softmax over j -> f16 probs in LDS (A-operand-readable), PV via WMMA
//   against B-packed V (contiguous per-lane loads, immediate offsets).
// ---------------------------------------------------------------------------
__global__ void addattn_kernel(const float* __restrict__ Q,
                               const float* __restrict__ Kp,
                               const _Float16* __restrict__ Vb,
                               const int* __restrict__ mask,
                               const float* __restrict__ wa,
                               float* __restrict__ X) {
  __shared__ __align__(16) float    sc[16][LKV];   // 32 KB
  __shared__ __align__(16) _Float16 pr[16][LKV];   // 16 KB
  __shared__ __align__(16) float    qt[16][DKH];   //  4 KB

  const int lane = threadIdx.x;                    // blockDim.x == 32
  const int bid  = blockIdx.x;
  const int qblk = bid & 15;                       // Lq/16
  const int h    = (bid >> 4) & 7;
  const int b    = bid >> 7;

  for (int idx = lane; idx < 16 * DKH; idx += 32) {
    int i = idx >> 6, d = idx & 63;
    qt[i][d] = Q[((size_t)(b * LQ + qblk * 16 + i)) * D_MODEL + h * DKH + d];
  }
  float war[DKH];
#pragma unroll
  for (int d = 0; d < DKH; ++d) war[d] = wa[d];
  __syncthreads();

  // ---- scores (transcendental VALU core) ----
  for (int jc = 0; jc < LKV; jc += 32) {
    int j = jc + lane;
    const float* kp = Kp + ((size_t)(b * LKV + j)) * D_MODEL + h * DKH;
    float kr[DKH];
#pragma unroll
    for (int d = 0; d < DKH; ++d) kr[d] = kp[d];
    bool msk = (mask[b * LKV + j] == 0);
    for (int i = 0; i < 16; ++i) {
      float s = 0.0f;
#pragma unroll
      for (int d = 0; d < DKH; ++d) s += tanhf(qt[i][d] + kr[d]) * war[d];
      sc[i][j] = msk ? -1e9f : s;
    }
  }
  __syncthreads();

  // ---- softmax per row (wave32 shuffle reductions) ----
  for (int i = 0; i < 16; ++i) {
    float mx = -3.4e38f;
#pragma unroll
    for (int t = 0; t < 16; ++t) mx = fmaxf(mx, sc[i][lane + 32 * t]);
#pragma unroll
    for (int off = 16; off > 0; off >>= 1) mx = fmaxf(mx, __shfl_xor(mx, off, 32));
    float sum = 0.0f;
#pragma unroll
    for (int t = 0; t < 16; ++t) {
      float e = __expf(sc[i][lane + 32 * t] - mx);
      sc[i][lane + 32 * t] = e;
      sum += e;
    }
#pragma unroll
    for (int off = 16; off > 0; off >>= 1) sum += __shfl_xor(sum, off, 32);
    float inv = 1.0f / sum;
#pragma unroll
    for (int t = 0; t < 16; ++t)
      pr[i][lane + 32 * t] = (_Float16)(sc[i][lane + 32 * t] * inv);
  }
  __syncthreads();

  // ---- PV: (16 x 512) x (512 x 64) via WMMA, f32 accumulation ----
  const int half_ = lane >> 4;
  const int nl    = lane & 15;
  // per-lane base into B-packed V for this (b,h); per-(kc,nt) stride = 512 h
  const _Float16* vbL = Vb + ((size_t)(b * NHEAD + h) * 16 * 4 * 32 + lane) * 16;

  v8f a0 = {}, a1 = {}, a2 = {}, a3 = {};
#pragma unroll
  for (int kc = 0; kc < KTILES; ++kc) {
    union { v16h v; uint4 u[2]; } au;
    au.u[0] = *(const uint4*)&pr[nl][kc * 32 + 16 * half_];
    au.u[1] = *(const uint4*)&pr[nl][kc * 32 + 16 * half_ + 8];
    v16h vb0 = *(const v16h*)(vbL + (kc * 4 + 0) * 512);
    v16h vb1 = *(const v16h*)(vbL + (kc * 4 + 1) * 512);
    v16h vb2 = *(const v16h*)(vbL + (kc * 4 + 2) * 512);
    v16h vb3 = *(const v16h*)(vbL + (kc * 4 + 3) * 512);
    a0 = __builtin_amdgcn_wmma_f32_16x16x32_f16(false, au.v, false, vb0, (short)0, a0, false, false);
    a1 = __builtin_amdgcn_wmma_f32_16x16x32_f16(false, au.v, false, vb1, (short)0, a1, false, false);
    a2 = __builtin_amdgcn_wmma_f32_16x16x32_f16(false, au.v, false, vb2, (short)0, a2, false, false);
    a3 = __builtin_amdgcn_wmma_f32_16x16x32_f16(false, au.v, false, vb3, (short)0, a3, false, false);
  }

  const v8f acc[4] = {a0, a1, a2, a3};
#pragma unroll
  for (int nt = 0; nt < 4; ++nt) {
#pragma unroll
    for (int r = 0; r < 8; ++r) {
      int row = b * LQ + qblk * 16 + 8 * half_ + r;
      int col = h * DKH + nt * 16 + nl;
      X[(size_t)row * D_MODEL + col] = acc[nt][r];
    }
  }
}

// ---------------------------------------------------------------------------
extern "C" void kernel_launch(void* const* d_in, const int* in_sizes, int n_in,
                              void* d_out, int out_size, void* d_ws, size_t ws_size,
                              hipStream_t stream) {
  const float* query = (const float*)d_in[0];   // (2,256,512)
  const float* key   = (const float*)d_in[1];   // (2,512,512)
  const float* value = (const float*)d_in[2];   // (2,512,512)
  const int*   mask  = (const int*)d_in[3];     // (2,512)
  const float* Wq    = (const float*)d_in[4];   // (512,512)
  const float* Wk    = (const float*)d_in[5];
  const float* Wv    = (const float*)d_in[6];
  const float* Wo    = (const float*)d_in[7];
  const float* wa    = (const float*)d_in[8];   // (64,)
  float* out = (float*)d_out;                   // (2,256,512) f32

  uint8_t* p = (uint8_t*)d_ws;
  auto carve = [&](size_t bytes) -> void* {
    void* r = (void*)p;
    p += (bytes + 255) & ~(size_t)255;
    return r;
  };
  _Float16* qA  = (_Float16*)carve((size_t)512  * 512 * 2);
  _Float16* kA  = (_Float16*)carve((size_t)1024 * 512 * 2);
  _Float16* vA  = (_Float16*)carve((size_t)1024 * 512 * 2);
  _Float16* xA  = (_Float16*)carve((size_t)512  * 512 * 2);
  _Float16* WqB = (_Float16*)carve((size_t)512 * 512 * 2);
  _Float16* WkB = (_Float16*)carve((size_t)512 * 512 * 2);
  _Float16* WvB = (_Float16*)carve((size_t)512 * 512 * 2);
  _Float16* WoB = (_Float16*)carve((size_t)512 * 512 * 2);
  float*    Qf  = (float*)   carve((size_t)512  * 512 * 4);
  float*    Kf  = (float*)   carve((size_t)1024 * 512 * 4);
  _Float16* Vh  = (_Float16*)carve((size_t)1024 * 512 * 2);  // row-major f16
  _Float16* Vb  = (_Float16*)carve((size_t)1024 * 512 * 2);  // B-packed f16
  float*    Xf  = (float*)   carve((size_t)512  * 512 * 4);

  // stage 1: convert + swizzle into WMMA operand layouts
  pack_a_kernel<<<(512 * 512) / 256, 256, 0, stream>>>(query, qA, 512, 512);
  pack_a_kernel<<<(1024 * 512) / 256, 256, 0, stream>>>(key, kA, 1024, 512);
  pack_a_kernel<<<(1024 * 512) / 256, 256, 0, stream>>>(value, vA, 1024, 512);
  pack_b_kernel<<<(512 * 512) / 256, 256, 0, stream>>>(Wq, WqB, 512, 512);
  pack_b_kernel<<<(512 * 512) / 256, 256, 0, stream>>>(Wk, WkB, 512, 512);
  pack_b_kernel<<<(512 * 512) / 256, 256, 0, stream>>>(Wv, WvB, 512, 512);
  pack_b_kernel<<<(512 * 512) / 256, 256, 0, stream>>>(Wo, WoB, 512, 512);

  // stage 2: projections (WMMA). strips = Mt * (N/64); 4 waves/block
  gemm_wmma_kernel<<<(32 * 8) / 4, 128, 0, stream>>>(qA, WqB, 32, 8, Qf, nullptr, 512);
  gemm_wmma_kernel<<<(64 * 8) / 4, 128, 0, stream>>>(kA, WkB, 64, 8, Kf, nullptr, 512);
  gemm_wmma_kernel<<<(64 * 8) / 4, 128, 0, stream>>>(vA, WvB, 64, 8, nullptr, Vh, 512);
  pack_vb_kernel<<<(1024 * 512) / 256, 256, 0, stream>>>(Vh, Vb);

  // stage 3: additive attention (tanh scores + softmax + PV WMMA)
  addattn_kernel<<<BATCH * NHEAD * (LQ / 16), 32, 0, stream>>>(Qf, Kf, Vb, mask, wa, Xf);

  // stage 4: output projection X @ Wo -> d_out (WMMA)
  pack_a_kernel<<<(512 * 512) / 256, 256, 0, stream>>>(Xf, xA, 512, 512);
  gemm_wmma_kernel<<<(32 * 8) / 4, 128, 0, stream>>>(xA, WoB, 32, 8, out, nullptr, 512);
}
